// DynamicMaskHead_2491081031723
// MI455X (gfx1250) — matrix operand order repaired and checked by reading, not compiled, not run on hardware
//
#include <hip/hip_runtime.h>

typedef __attribute__((ext_vector_type(16))) _Float16 v16h;
typedef __attribute__((ext_vector_type(8)))  float    v8f;

#define HH 128        // logits height
#define WW 192        // logits width
#define HW (HH*WW)
#define TH 8          // logits rows per workgroup band
#define NP 169        // params per instance: 80 w0 + 64 w1 + 8 w2 + 8 b0 + 8 b1 + 1 b2

// One workgroup = one (instance, 8-row band).
// Phase 1: 8 waves compute 9 logits rows (8 + halo) into LDS via chained f16 WMMA.
// Phase 2: 2x aligned-bilinear upsample from LDS -> global out (written once).
__global__ __launch_bounds__(256)
void dynmask_fused(const float* __restrict__ mask_feats,   // (4,8,128,192)
                   const float* __restrict__ params,       // (n_inst,169)
                   const float* __restrict__ locs,         // (n_inst,2)
                   const float* __restrict__ soi,          // (5)
                   const int*   __restrict__ im_inds,      // (n_inst)
                   const int*   __restrict__ fpn_levels,   // (n_inst)
                   const int*   __restrict__ stride_p,     // scalar (8)
                   float*       __restrict__ out)          // (n_inst,1,256,384)
{
    __shared__ float s_logit[(TH + 1) * WW];   // 6912 B

    const int blk    = blockIdx.x;
    const int inst   = blk >> 4;               // 16 bands per instance (128/8)
    const int rb     = blk & 15;
    const int y0     = rb * TH;

    const int tid    = threadIdx.x;
    const int wave   = tid >> 5;
    const int lane   = tid & 31;
    const int lane15 = lane & 15;
    const bool hi    = lane >= 16;

    const float stride = (float)stride_p[0];
    const float halfs  = 0.5f * stride;

    const int   im      = im_inds[inst];
    const float inv_soi = 1.0f / soi[fpn_levels[inst]];
    const float locx    = locs[2 * inst + 0];
    const float locy    = locs[2 * inst + 1];
    const float* p      = params + inst * NP;

    // ---- Loop-invariant A matrices (weights) and C matrices (biases) ----
    // K mapping for layer 0: K=0..7 -> feature channels (w0[m][2+k]),
    //                        K=8 -> rel-x (w0[m][0]), K=9 -> rel-y (w0[m][1]).
    // A layout: lanes 0-15 = rows M (K 0-7 in elems 0-7, K 16-23 in elems 8-15);
    //           lanes 16-31 = same rows (K 8-15 in elems 0-7, K 24-31 in 8-15).
    v16h a0 = {}; v16h a1 = {}; v16h a2 = {};
    v8f  c0 = {}; v8f  c1 = {}; v8f  c2 = {};
    {
        const int m = lane15;
        if (!hi) {
            if (m < 8) {
                #pragma unroll
                for (int e = 0; e < 8; ++e) {
                    a0[e] = (_Float16)p[m * 10 + 2 + e];     // w0 feature weights
                    a1[e] = (_Float16)p[80 + m * 8 + e];     // w1
                }
            }
            if (lane == 0) {
                #pragma unroll
                for (int e = 0; e < 8; ++e) a2[e] = (_Float16)p[144 + e];  // w2 row 0
            }
            // C/D layout: lane (0-15) = column, VGPR v = row v  -> bias per row.
            #pragma unroll
            for (int v = 0; v < 8; ++v) { c0[v] = p[152 + v]; c1[v] = p[160 + v]; }
            c2[0] = p[168];
        } else {
            if (m < 8) {
                a0[0] = (_Float16)p[m * 10 + 0];   // K=8: rel-x weight
                a0[1] = (_Float16)p[m * 10 + 1];   // K=9: rel-y weight
            }
        }
    }

    // ---- Phase 1: logits for (TH+1) rows, 16 pixels per WMMA group ----
    const int groups = (TH + 1) * WW / 16;   // 108; 16 | 192 so a group never crosses a row
    for (int g = wave; g < groups; g += 8) {
        const int lpix = g * 16 + lane15;        // local pixel index in band
        const int ly   = lpix / WW;              // 0..TH (row TH = halo)
        const int lx   = lpix - ly * WW;
        const int gy   = min(y0 + ly, HH - 1);   // edge clamp == reference edge pad

        // B layout: lane%16 = column (pixel); lanes 0-15 elems 0-7 = K 0-7, lanes 16-31 elems 0-7 = K 8-15.
        v16h b0 = {};
        if (!hi) {
            const float* f = mask_feats + (size_t)im * 8 * HW + gy * WW + lx;
            #pragma unroll
            for (int c = 0; c < 8; ++c) b0[c] = (_Float16)f[c * HW];   // coalesced 64B/ch
        } else {
            b0[0] = (_Float16)((locx - (lx * stride + halfs)) * inv_soi);
            b0[1] = (_Float16)((locy - (gy * stride + halfs)) * inv_soi);
        }

        v8f d0 = __builtin_amdgcn_wmma_f32_16x16x32_f16(
            false, a0, false, b0, (short)0, c0, false, false);

        // relu + pack: D (channels x pixels) is already in next layer's B layout.
        v16h h1 = {};
        #pragma unroll
        for (int v = 0; v < 8; ++v) h1[v] = (_Float16)fmaxf(d0[v], 0.0f);

        v8f d1 = __builtin_amdgcn_wmma_f32_16x16x32_f16(
            false, a1, false, h1, (short)0, c1, false, false);

        v16h h2 = {};
        #pragma unroll
        for (int v = 0; v < 8; ++v) h2[v] = (_Float16)fmaxf(d1[v], 0.0f);

        v8f d2 = __builtin_amdgcn_wmma_f32_16x16x32_f16(
            false, a2, false, h2, (short)0, c2, false, false);

        if (!hi) s_logit[lpix] = d2[0];          // row 0 of D2 = logits of 16 pixels
    }

    __syncthreads();

    // ---- Phase 2: 2x aligned bilinear upsample from LDS ----
    // out[i][j] = bilerp of logits at (max(i-1,0)/2, max(j-1,0)/2), edge-clamped.
    const int OW      = 2 * WW;                       // 384
    const int i_start = (rb == 0) ? 0 : (2 * y0 + 1);
    const int i_end   = min(2 * y0 + 2 * TH, 2 * HH - 1);
    const int nrows   = i_end - i_start + 1;
    float* obase = out + (size_t)inst * (2 * HH) * OW;

    for (int t = tid; t < nrows * OW; t += 256) {
        const int i  = i_start + t / OW;
        const int j  = t - (t / OW) * OW;
        const int r  = (i > 0) ? (i - 1) : 0;
        const int c  = (j > 0) ? (j - 1) : 0;
        const int yl = (r >> 1) - y0;                 // local row in [0, TH]
        const int yh = yl + (r & 1);                  // halo row covers +1
        const int xl = c >> 1;
        const int xh = min(xl + (c & 1), WW - 1);
        const float wy = (r & 1) ? 0.5f : 0.0f;
        const float wx = (c & 1) ? 0.5f : 0.0f;

        const float v00 = s_logit[yl * WW + xl];
        const float v01 = s_logit[yl * WW + xh];
        const float v10 = s_logit[yh * WW + xl];
        const float v11 = s_logit[yh * WW + xh];
        const float top = v00 + (v01 - v00) * wx;
        const float bot = v10 + (v11 - v10) * wx;
        obase[(size_t)i * OW + j] = top + (bot - top) * wy;
    }
}

extern "C" void kernel_launch(void* const* d_in, const int* in_sizes, int n_in,
                              void* d_out, int out_size, void* d_ws, size_t ws_size,
                              hipStream_t stream) {
    (void)n_in; (void)out_size; (void)d_ws; (void)ws_size;
    const float* mask_feats = (const float*)d_in[0];
    const float* params     = (const float*)d_in[1];
    const float* locs       = (const float*)d_in[2];
    const float* soi        = (const float*)d_in[3];
    const int*   im_inds    = (const int*)  d_in[4];
    const int*   fpn_levels = (const int*)  d_in[5];
    const int*   stride_p   = (const int*)  d_in[6];
    float*       out        = (float*)d_out;

    const int n_inst = in_sizes[4];                  // 128
    dim3 grid(n_inst * (HH / TH));                   // 2048 workgroups
    dynmask_fused<<<grid, 256, 0, stream>>>(mask_feats, params, locs, soi,
                                            im_inds, fpn_levels, stride_p, out);
}
